// Attention_34677565948071
// MI455X (gfx1250) — matrix operand order repaired and checked by reading, not compiled, not run on hardware
//
#include <hip/hip_runtime.h>
#include <hip/hip_bf16.h>

// ---------------------------------------------------------------------------
// CDNA5 (gfx1250) sliding-window attention, bf16 WMMA (16x16x32) everywhere.
// Reference returns (out, attn): d_out = out[4,4096,1024] ++ attn[4,4096,4096]
// ---------------------------------------------------------------------------

typedef __bf16 bf16;
typedef __attribute__((ext_vector_type(16))) __bf16 v16bf;
typedef __attribute__((ext_vector_type(8)))  __bf16 v8bf;
typedef __attribute__((ext_vector_type(8)))  float  v8f;

#define NB   4
#define NS   4096
#define ND   1024      // DIN = DINNER = DOUT
#define NQKV 3072
#define WIN  99        // lookback (window N-1)

#define WMMA_BF16(a, b, c) \
  __builtin_amdgcn_wmma_f32_16x16x32_bf16(false, (a), false, (b), (short)0, (c), false, false)

__device__ __forceinline__ v16bf join16(v8bf lo, v8bf hi) {
  v16bf r;
#pragma unroll
  for (int e = 0; e < 8; ++e) { r[e] = lo[e]; r[e + 8] = hi[e]; }
  return r;
}

// ------------------------------ fp32 -> bf16 -------------------------------
__global__ void k_cvt(const float* __restrict__ s, bf16* __restrict__ d, int n) {
  int i = (blockIdx.x * 256 + threadIdx.x) * 8;
  if (i >= n) return;
  float4 f0 = *(const float4*)(s + i);
  float4 f1 = *(const float4*)(s + i + 4);
  v8bf h;
  h[0] = (bf16)f0.x; h[1] = (bf16)f0.y; h[2] = (bf16)f0.z; h[3] = (bf16)f0.w;
  h[4] = (bf16)f1.x; h[5] = (bf16)f1.y; h[6] = (bf16)f1.z; h[7] = (bf16)f1.w;
  *(v8bf*)(d + i) = h;
}

// ------------------------------- zero fill ---------------------------------
__global__ void k_zero(float4* __restrict__ p, int n4) {
  int i = blockIdx.x * 256 + threadIdx.x;
  if (i < n4) {
    float4 z; z.x = 0.f; z.y = 0.f; z.z = 0.f; z.w = 0.f;
    p[i] = z;
  }
}

// -------------------- GEMM1: qkv = x @ W_qkv  (bf16 WMMA) ------------------
// x: [16384,1024] f32 (converted to bf16 while staging to LDS)
// W_qkv (bf16): [1024,3072].  Outputs: Q,K row-major bf16 [16384,1024],
// V transposed bf16 vT[b][d][s] so PV B-fragments are contiguous later.
__global__ __launch_bounds__(256) void k_gemm_qkv(
    const float* __restrict__ X, const bf16* __restrict__ W,
    bf16* __restrict__ Q, bf16* __restrict__ K, bf16* __restrict__ VT) {
  constexpr int LDA = 48;                 // 32 + 16 pad -> 96B stride, 16B aligned
  __shared__ bf16 As[128 * LDA];
  __shared__ bf16 Bs[128 * LDA];          // transposed: Bs[n][k]
  const int tid = threadIdx.x;
  const int wave = tid >> 5, lane = tid & 31;
  const int khalf = lane >> 4, l16 = lane & 15;
  const int m0 = blockIdx.x * 128;
  const int n0 = blockIdx.y * 128;
  const int wm = (wave >> 2) * 64;        // 2 wave rows of 64
  const int wn = (wave & 3) * 32;         // 4 wave cols of 32

  v8f acc[4][2];
#pragma unroll
  for (int i = 0; i < 4; ++i)
#pragma unroll
    for (int j = 0; j < 2; ++j)
#pragma unroll
      for (int e = 0; e < 8; ++e) acc[i][j][e] = 0.f;

  for (int k0 = 0; k0 < ND; k0 += 32) {
    {   // stage A (f32 -> bf16): each thread = 16 contiguous elems of one row
      int row = tid >> 1, seg = (tid & 1) * 16;
      const float* src = X + (size_t)(m0 + row) * ND + k0 + seg;
      float4 f0 = *(const float4*)(src);
      float4 f1 = *(const float4*)(src + 4);
      float4 f2 = *(const float4*)(src + 8);
      float4 f3 = *(const float4*)(src + 12);
      v8bf h0, h1;
      h0[0]=(bf16)f0.x; h0[1]=(bf16)f0.y; h0[2]=(bf16)f0.z; h0[3]=(bf16)f0.w;
      h0[4]=(bf16)f1.x; h0[5]=(bf16)f1.y; h0[6]=(bf16)f1.z; h0[7]=(bf16)f1.w;
      h1[0]=(bf16)f2.x; h1[1]=(bf16)f2.y; h1[2]=(bf16)f2.z; h1[3]=(bf16)f2.w;
      h1[4]=(bf16)f3.x; h1[5]=(bf16)f3.y; h1[6]=(bf16)f3.z; h1[7]=(bf16)f3.w;
      *(v8bf*)(As + row * LDA + seg) = h0;
      *(v8bf*)(As + row * LDA + seg + 8) = h1;
    }
    {   // stage B transposed: thread = 2 k-rows x 8 n; pack k-pairs -> b32
      int kp = (tid >> 4) * 2;            // k-row pair 0,2,..,30
      int seg = (tid & 15) * 8;           // 8 n per thread
      const bf16* s0 = W + (size_t)(k0 + kp) * NQKV + n0 + seg;
      const bf16* s1 = s0 + NQKV;
      v8bf w0 = *(const v8bf*)s0;
      v8bf w1 = *(const v8bf*)s1;
#pragma unroll
      for (int q = 0; q < 8; ++q) {
        v8bf pr;                          // only [0],[1] used; store as b32
        pr[0] = w0[q]; pr[1] = w1[q];
        *(uint*)(Bs + (seg + q) * LDA + kp) = *(uint*)&pr;
      }
    }
    __syncthreads();

    v16bf af[4], bfr[2];
#pragma unroll
    for (int mi = 0; mi < 4; ++mi) {
      const bf16* p = As + (wm + mi * 16 + l16) * LDA + khalf * 16;
      af[mi] = join16(*(const v8bf*)p, *(const v8bf*)(p + 8));
    }
#pragma unroll
    for (int ni = 0; ni < 2; ++ni) {
      const bf16* p = Bs + (wn + ni * 16 + l16) * LDA + khalf * 16;
      bfr[ni] = join16(*(const v8bf*)p, *(const v8bf*)(p + 8));
    }
#pragma unroll
    for (int mi = 0; mi < 4; ++mi)
#pragma unroll
      for (int ni = 0; ni < 2; ++ni)
        acc[mi][ni] = WMMA_BF16(af[mi], bfr[ni], acc[mi][ni]);
    __syncthreads();
  }

  // Branch-free epilogue: uniform region -> pointer + per-r stride chosen once.
  const int region = n0 >> 10;            // 0=q 1=k 2=v (N-tile never straddles)
  bf16* dstp = (region == 0) ? Q : (region == 1) ? K : VT;
  const size_t step = (region == 2) ? 1 : (size_t)ND;
#pragma unroll
  for (int mi = 0; mi < 4; ++mi)
#pragma unroll
    for (int ni = 0; ni < 2; ++ni) {
      int gm0 = m0 + wm + mi * 16 + khalf * 8;        // 8-row group, one batch
      int gn  = (n0 & 1023) + wn + ni * 16 + l16;
      size_t base;
      if (region == 2) {
        int b = gm0 >> 12, s = gm0 & 4095;
        base = ((size_t)b * ND + gn) * NS + s;        // vT[b][d][s], step 1 in s
      } else {
        base = (size_t)gm0 * ND + gn;                 // row-major, step ND
      }
#pragma unroll
      for (int r = 0; r < 8; ++r)
        dstp[base + (size_t)r * step] = (bf16)acc[mi][ni][r];
    }
}

// --------------- windowed softmax: attn = softmax(mask(q k^T)) -------------
// One wave per 16-row tile; j covered by <=5 tiles of 32 columns.
__global__ __launch_bounds__(256) void k_attn_softmax(
    const bf16* __restrict__ Q, const bf16* __restrict__ Kk,
    float* __restrict__ ATTN) {
  const int tid = threadIdx.x, wave = tid >> 5, lane = tid & 31;
  const int khalf = lane >> 4, l16 = lane & 15;
  const int tile = blockIdx.x * 8 + wave;      // 1024 tiles total
  const int b = tile >> 8;
  const int i0 = (tile & 255) << 4;
  const int ilast = i0 + 15;
  int jlo = i0 - WIN; if (jlo < 0) jlo = 0;
  const int jt0 = jlo & ~31;

  v8f sacc[5][2];
#pragma unroll
  for (int t = 0; t < 5; ++t)
#pragma unroll
    for (int c = 0; c < 2; ++c)
#pragma unroll
      for (int e = 0; e < 8; ++e) sacc[t][c][e] = 0.f;

  const bf16* qrow = Q + ((size_t)(b * NS + i0 + l16)) * ND;
  for (int d0 = 0; d0 < ND; d0 += 32) {
    v16bf aq = join16(*(const v8bf*)(qrow + d0 + khalf * 16),
                      *(const v8bf*)(qrow + d0 + khalf * 16 + 8));
#pragma unroll
    for (int t = 0; t < 5; ++t) {
      int j0 = jt0 + t * 32;
      if (j0 <= ilast) {
#pragma unroll
        for (int cs = 0; cs < 2; ++cs) {
          int j = j0 + cs * 16 + l16;
          const bf16* kr = Kk + ((size_t)(b * NS + j)) * ND + d0 + khalf * 16;
          v16bf bk = join16(*(const v8bf*)kr, *(const v8bf*)(kr + 8));
          sacc[t][cs] = WMMA_BF16(aq, bk, sacc[t][cs]);
        }
      }
    }
  }

  const float scale = 0.03125f;           // 1024^-0.5
  float mrow[8], srow[8];
#pragma unroll
  for (int r = 0; r < 8; ++r) mrow[r] = -3.0e38f;
#pragma unroll
  for (int t = 0; t < 5; ++t) {
    int j0 = jt0 + t * 32;
    if (j0 <= ilast) {
#pragma unroll
      for (int cs = 0; cs < 2; ++cs)
#pragma unroll
        for (int r = 0; r < 8; ++r) {
          int i = i0 + khalf * 8 + r;
          int j = j0 + cs * 16 + l16;
          float v = sacc[t][cs][r] * scale;
          bool ok = (j <= i) && (j >= i - WIN);
          v = ok ? v : -3.0e38f;
          sacc[t][cs][r] = v;
          mrow[r] = fmaxf(mrow[r], v);
        }
    }
  }
#pragma unroll
  for (int r = 0; r < 8; ++r) {
#pragma unroll
    for (int m = 1; m < 16; m <<= 1)
      mrow[r] = fmaxf(mrow[r], __shfl_xor(mrow[r], m, 32));
    srow[r] = 0.f;
  }
#pragma unroll
  for (int t = 0; t < 5; ++t) {
    int j0 = jt0 + t * 32;
    if (j0 <= ilast) {
#pragma unroll
      for (int cs = 0; cs < 2; ++cs)
#pragma unroll
        for (int r = 0; r < 8; ++r) {
          float p = __expf(sacc[t][cs][r] - mrow[r]);
          sacc[t][cs][r] = p;
          srow[r] += p;
        }
    }
  }
#pragma unroll
  for (int r = 0; r < 8; ++r) {
#pragma unroll
    for (int m = 1; m < 16; m <<= 1)
      srow[r] += __shfl_xor(srow[r], m, 32);
    srow[r] = 1.f / srow[r];
  }
  float* ab = ATTN + (size_t)b * NS * NS;
#pragma unroll
  for (int t = 0; t < 5; ++t) {
    int j0 = jt0 + t * 32;
    if (j0 <= ilast) {
#pragma unroll
      for (int cs = 0; cs < 2; ++cs)
#pragma unroll
        for (int r = 0; r < 8; ++r) {
          int i = i0 + khalf * 8 + r;
          int j = j0 + cs * 16 + l16;
          ab[(size_t)i * NS + j] = sacc[t][cs][r] * srow[r];
        }
    }
  }
}

// ----------------------- banded t = attn @ v  (WMMA) -----------------------
// A from attn (f32, band is exactly the tiles written above; rest is 0).
// B from vT[b][d][s]: lane=column d, 16 consecutive j -> contiguous loads.
__global__ __launch_bounds__(128) void k_attn_v(
    const float* __restrict__ ATTN, const bf16* __restrict__ VT,
    bf16* __restrict__ T) {
  const int tid = threadIdx.x, wave = tid >> 5, lane = tid & 31;
  const int khalf = lane >> 4, l16 = lane & 15;
  const int rt = blockIdx.x;               // 1024 row tiles
  const int b = rt >> 8;
  const int i0 = (rt & 255) << 4;
  const int nb = blockIdx.y * 256 + wave * 64;
  const int ilast = i0 + 15;
  int jlo = i0 - WIN; if (jlo < 0) jlo = 0;
  const int jt0 = jlo & ~31;

  v8f acc[4];
#pragma unroll
  for (int i = 0; i < 4; ++i)
#pragma unroll
    for (int e = 0; e < 8; ++e) acc[i][e] = 0.f;

  const float* arow = ATTN + ((size_t)(b * NS + i0 + l16)) * NS;
#pragma unroll
  for (int t = 0; t < 5; ++t) {
    int j0 = jt0 + t * 32;
    if (j0 <= ilast) {
      const float* p0 = arow + j0 + khalf * 16;
      float4 f0 = *(const float4*)(p0);
      float4 f1 = *(const float4*)(p0 + 4);
      float4 f2 = *(const float4*)(p0 + 8);
      float4 f3 = *(const float4*)(p0 + 12);
      v16bf ap;
      ap[0]=(bf16)f0.x; ap[1]=(bf16)f0.y; ap[2]=(bf16)f0.z; ap[3]=(bf16)f0.w;
      ap[4]=(bf16)f1.x; ap[5]=(bf16)f1.y; ap[6]=(bf16)f1.z; ap[7]=(bf16)f1.w;
      ap[8]=(bf16)f2.x; ap[9]=(bf16)f2.y; ap[10]=(bf16)f2.z; ap[11]=(bf16)f2.w;
      ap[12]=(bf16)f3.x; ap[13]=(bf16)f3.y; ap[14]=(bf16)f3.z; ap[15]=(bf16)f3.w;
#pragma unroll
      for (int ni = 0; ni < 4; ++ni) {
        int d = nb + ni * 16 + l16;
        const bf16* vp = VT + ((size_t)b * ND + d) * NS + j0 + khalf * 16;
        v16bf bv = join16(*(const v8bf*)vp, *(const v8bf*)(vp + 8));
        acc[ni] = WMMA_BF16(ap, bv, acc[ni]);
      }
    }
  }
#pragma unroll
  for (int ni = 0; ni < 4; ++ni)
#pragma unroll
    for (int r = 0; r < 8; ++r)
      T[((size_t)(b * NS + i0 + khalf * 8 + r)) * ND + nb + ni * 16 + l16] =
          (bf16)acc[ni][r];
}

// ------------------- GEMM2: out = t @ W_out + b_out (f32) ------------------
__global__ __launch_bounds__(256) void k_gemm_out(
    const bf16* __restrict__ T, const bf16* __restrict__ W,
    const float* __restrict__ bias, float* __restrict__ OUT) {
  constexpr int LDA = 48;
  __shared__ bf16 As[128 * LDA];
  __shared__ bf16 Bs[128 * LDA];
  const int tid = threadIdx.x;
  const int wave = tid >> 5, lane = tid & 31;
  const int khalf = lane >> 4, l16 = lane & 15;
  const int m0 = blockIdx.x * 128;
  const int n0 = blockIdx.y * 128;
  const int wm = (wave >> 2) * 64;
  const int wn = (wave & 3) * 32;

  v8f acc[4][2];
#pragma unroll
  for (int i = 0; i < 4; ++i)
#pragma unroll
    for (int j = 0; j < 2; ++j)
#pragma unroll
      for (int e = 0; e < 8; ++e) acc[i][j][e] = 0.f;

  for (int k0 = 0; k0 < ND; k0 += 32) {
    {
      int row = tid >> 1, seg = (tid & 1) * 16;
      const bf16* src = T + (size_t)(m0 + row) * ND + k0 + seg;
      *(v8bf*)(As + row * LDA + seg) = *(const v8bf*)src;
      *(v8bf*)(As + row * LDA + seg + 8) = *(const v8bf*)(src + 8);
    }
    {
      int kp = (tid >> 4) * 2;
      int seg = (tid & 15) * 8;
      const bf16* s0 = W + (size_t)(k0 + kp) * ND + n0 + seg;
      const bf16* s1 = s0 + ND;
      v8bf w0 = *(const v8bf*)s0;
      v8bf w1 = *(const v8bf*)s1;
#pragma unroll
      for (int q = 0; q < 8; ++q) {
        v8bf pr;
        pr[0] = w0[q]; pr[1] = w1[q];
        *(uint*)(Bs + (seg + q) * LDA + kp) = *(uint*)&pr;
      }
    }
    __syncthreads();

    v16bf af[4], bfr[2];
#pragma unroll
    for (int mi = 0; mi < 4; ++mi) {
      const bf16* p = As + (wm + mi * 16 + l16) * LDA + khalf * 16;
      af[mi] = join16(*(const v8bf*)p, *(const v8bf*)(p + 8));
    }
#pragma unroll
    for (int ni = 0; ni < 2; ++ni) {
      const bf16* p = Bs + (wn + ni * 16 + l16) * LDA + khalf * 16;
      bfr[ni] = join16(*(const v8bf*)p, *(const v8bf*)(p + 8));
    }
#pragma unroll
    for (int mi = 0; mi < 4; ++mi)
#pragma unroll
      for (int ni = 0; ni < 2; ++ni)
        acc[mi][ni] = WMMA_BF16(af[mi], bfr[ni], acc[mi][ni]);
    __syncthreads();
  }

#pragma unroll
  for (int mi = 0; mi < 4; ++mi)
#pragma unroll
    for (int ni = 0; ni < 2; ++ni) {
      int gn = n0 + wn + ni * 16 + l16;
      float bb = bias[gn];
#pragma unroll
      for (int r = 0; r < 8; ++r) {
        int gm = m0 + wm + mi * 16 + khalf * 8 + r;
        OUT[(size_t)gm * ND + gn] = acc[mi][ni][r] + bb;
      }
    }
}

// ---------------------------------------------------------------------------
extern "C" void kernel_launch(void* const* d_in, const int* in_sizes, int n_in,
                              void* d_out, int out_size, void* d_ws, size_t ws_size,
                              hipStream_t stream) {
  const float* x    = (const float*)d_in[0];   // [4,4096,1024]
  const float* wqkv = (const float*)d_in[1];   // [1024,3072]
  const float* wout = (const float*)d_in[2];   // [1024,1024]
  const float* bout = (const float*)d_in[3];   // [1024]

  float* out  = (float*)d_out;                          // 16,777,216 f32
  float* attn = out + (size_t)NB * NS * ND;             // 67,108,864 f32

  // workspace layout (bf16)
  bf16* Wqb = (bf16*)d_ws;                              // 1024*3072
  bf16* Wob = Wqb + (size_t)ND * NQKV;                  // 1024*1024
  bf16* Q   = Wob + (size_t)ND * ND;                    // 16384*1024
  bf16* K   = Q  + (size_t)NB * NS * ND;
  bf16* VT  = K  + (size_t)NB * NS * ND;                // [b][d][s]
  bf16* T   = VT + (size_t)NB * NS * ND;                // attn@v result

  // 1) convert weights to bf16
  k_cvt<<<(ND * NQKV) / 2048, 256, 0, stream>>>(wqkv, Wqb, ND * NQKV);
  k_cvt<<<(ND * ND) / 2048, 256, 0, stream>>>(wout, Wob, ND * ND);

  // 2) zero the dense attn output (band is overwritten later)
  k_zero<<<(NB * NS * NS / 4) / 256, 256, 0, stream>>>((float4*)attn,
                                                       NB * NS * NS / 4);

  // 3) qkv projection GEMM (writes Q, K row-major; V transposed)
  k_gemm_qkv<<<dim3(128, 24), 256, 0, stream>>>(x, Wqb, Q, K, VT);

  // 4) windowed softmax -> attn output region
  k_attn_softmax<<<128, 256, 0, stream>>>(Q, K, attn);

  // 5) banded attn @ v -> T (bf16)
  k_attn_v<<<dim3(1024, 4), 128, 0, stream>>>(attn, VT, T);

  // 6) output projection + bias -> out (f32)
  k_gemm_out<<<dim3(128, 8), 256, 0, stream>>>(T, Wob, bout, out);
}
